// WARM_80418967650697
// MI455X (gfx1250) — matrix operand order repaired and checked by reading, not compiled, not run on hardware
//
#include <hip/hip_runtime.h>
#include <hip/hip_bf16.h>

typedef __bf16 v16bf __attribute__((ext_vector_type(16)));
typedef float  v8f   __attribute__((ext_vector_type(8)));

#define BDIM  8
#define CDIM  64
#define HDIM  256
#define WDIM  256
#define H2    128
#define W2    128
#define NG    4
#define PTILE 32   // half-res pixels per workgroup
#define CSTR  65   // padded channel stride in LDS (floats, odd -> conflict-free)

// ---- workspace layout (prepped once per launch) ----
// [0, 64KB)      : 64 A-fragments x 32 lanes x 16 bf16 (32B/lane, WMMA A layout)
//                  frag id f = ((s*2 + mat)*4 + mt)*2 + kb   (mat: 0=Wg, 1=Wx)
// [64KB, +1KB)   : fused bias  bsum[s][64] = bg+bx  (f32)
#define WS_FRAG_BYTES (64 * 32 * 16 * 2)
#define WS_BIAS_OFF   WS_FRAG_BYTES
#define WS_TOTAL      (WS_FRAG_BYTES + NG * CDIM * 4)

__global__ __launch_bounds__(256) void prep_weights(
    const float* __restrict__ Wg, const float* __restrict__ Wx,
    const float* __restrict__ bg, const float* __restrict__ bx,
    void* __restrict__ ws)
{
  __bf16* frag = (__bf16*)ws;
  float*  bsum = (float*)((char*)ws + WS_BIAS_OFF);
  const int tid  = blockIdx.x * blockDim.x + threadIdx.x;   // 2048 = 64 frags x 32 lanes
  const int lane = tid & 31;
  const int f    = tid >> 5;
  const int kb   = f & 1;
  const int mt   = (f >> 1) & 3;
  const int mat  = (f >> 3) & 1;
  const int s    = f >> 4;
  const float* Wsrc = (mat == 0 ? Wg : Wx) + s * CDIM * CDIM;
  const int row = mt * 16 + (lane & 15);            // A layout: lane m / m+16 hold row m
  const int ks  = kb * 32 + (lane >> 4) * 8;        // lane<16: K {0..7,16..23}; hi: +8
  __bf16* dst = frag + (size_t)tid * 16;
  #pragma unroll
  for (int e = 0; e < 16; ++e) {
    const int k = ks + ((e < 8) ? e : (e + 8));
    dst[e] = (__bf16)Wsrc[row * CDIM + k];
  }
  if (f < 8) {                                      // threads 0..255 -> 256 bias entries
    const int o  = f * 32 + lane;
    const int ss = o >> 6, oo = o & 63;
    bsum[o] = bg[ss * CDIM + oo] + bx[ss * CDIM + oo];
  }
}

template <bool USE_WS>
__global__ __launch_bounds__(256) void wavelet_attn_gate_fused(
    const float* __restrict__ xe,  const float* __restrict__ xd,
    const float* __restrict__ Wg,  const float* __restrict__ bg,
    const float* __restrict__ Wx,  const float* __restrict__ bx,
    const float* __restrict__ Wps, const float* __restrict__ bps,
    const void* __restrict__ ws,   float* __restrict__ out)
{
  __shared__ float sSub[NG][PTILE][CSTR];   // fp32 subbands [s][pixel][channel]
  __shared__ float sG[PTILE][CSTR];         // fp32 pooled gate
  __shared__ float sAtt[NG][PTILE];         // per-pixel attention scalars

  const int tid = threadIdx.x;
  const int blk = blockIdx.x;
  const int q   = blk & 3;                  // quarter of a half-res row (32 px)
  const int h2  = (blk >> 2) & (H2 - 1);    // half-res row
  const int b   = blk >> 9;                 // batch
  const int w0  = q * (2 * PTILE);          // full-res column base

  // -------- Phase 1: stream in (b128 pixel pairs), Haar DWT + 2x2 mean pool -----
  #pragma unroll
  for (int i = 0; i < 4; ++i) {
    const int idx = i * 256 + tid;          // 64 ch x 16 pixel-pairs
    const int c  = idx >> 4;
    const int pp = idx & 15;                // pixel pair -> half-res pixels 2pp, 2pp+1
    const long base = (((long)(b * CDIM + c) * HDIM + 2 * h2) * WDIM) + (w0 + 4 * pp);
    const float4 e0 = *reinterpret_cast<const float4*>(xe + base);
    const float4 e1 = *reinterpret_cast<const float4*>(xe + base + WDIM);
    const float4 d0 = *reinterpret_cast<const float4*>(xd + base);
    const float4 d1 = *reinterpret_cast<const float4*>(xd + base + WDIM);
    if (i < 3) {                            // emits global_prefetch_b8
      const long nb = base + 16L * HDIM * WDIM;
      __builtin_prefetch(xe + nb, 0, 1);
      __builtin_prefetch(xd + nb, 0, 1);
    }
    const int p0 = 2 * pp, p1 = 2 * pp + 1;
    sSub[0][p0][c] = (e0.x + e0.y + e1.x + e1.y) * 0.5f;
    sSub[1][p0][c] = (e0.x + e0.y - e1.x - e1.y) * 0.5f;
    sSub[2][p0][c] = (e0.x - e0.y + e1.x - e1.y) * 0.5f;
    sSub[3][p0][c] = (e0.x - e0.y - e1.x + e1.y) * 0.5f;
    sG[p0][c]      = (d0.x + d0.y + d1.x + d1.y) * 0.25f;
    sSub[0][p1][c] = (e0.z + e0.w + e1.z + e1.w) * 0.5f;
    sSub[1][p1][c] = (e0.z + e0.w - e1.z - e1.w) * 0.5f;
    sSub[2][p1][c] = (e0.z - e0.w + e1.z - e1.w) * 0.5f;
    sSub[3][p1][c] = (e0.z - e0.w - e1.z + e1.w) * 0.5f;
    sG[p1][c]      = (d0.z + d0.w + d1.z + d1.w) * 0.25f;
  }
  __syncthreads();

  // -------- Phase 2: per-wave (subband, pixel-tile) WMMA task --------------------
  const int wave = tid >> 5;
  const int lane = tid & 31;
  const int s    = wave >> 1;               // subband 0..3
  const int t    = wave & 1;                // N-tile 0..1
  const int hi   = lane >> 4;               // lane half
  const int pn   = t * 16 + (lane & 15);    // this lane's pixel column (N)

  // B fragments (K = channels): lanes 0-15 hold K {0..7,16..23} (+kb*32), lanes 16-31 +8
  v16bf bGf[2], bXf[2];
  #pragma unroll
  for (int kb = 0; kb < 2; ++kb) {
    const int cb = kb * 32 + hi * 8;
    const float* gsrc = &sG[pn][cb];
    const float* xsrc = &sSub[s][pn][cb];
    #pragma unroll
    for (int e = 0; e < 16; ++e) {
      bGf[kb][e] = (__bf16)gsrc[(e < 8) ? e : (e + 8)];
      bXf[kb][e] = (__bf16)xsrc[(e < 8) ? e : (e + 8)];
    }
  }

  const __bf16* frag = (const __bf16*)ws;
  const float*  bsum = (const float*)((const char*)ws + WS_BIAS_OFF);
  const float* Wg_s = Wg  + s * CDIM * CDIM;
  const float* Wx_s = Wx  + s * CDIM * CDIM;
  const float* bg_s = bg  + s * CDIM;
  const float* bx_s = bx  + s * CDIM;
  const float* wp_s = Wps + s * CDIM;

  float S = 0.0f;  // partial psi dot-product for this lane's pixel
  #pragma unroll
  for (int mt = 0; mt < 4; ++mt) {
    v8f acc;
    #pragma unroll
    for (int r = 0; r < 8; ++r) {           // C/D layout: lane<16 -> M=r, hi -> M=r+8
      const int o = mt * 16 + hi * 8 + r;
      acc[r] = USE_WS ? bsum[s * CDIM + o] : (bg_s[o] + bx_s[o]);
    }
    #pragma unroll
    for (int kb = 0; kb < 2; ++kb) {
      v16bf aG, aX;
      if (USE_WS) {
        // pre-swizzled bf16 fragments: one contiguous 32B load per lane
        const int fG = ((s * 2 + 0) * 4 + mt) * 2 + kb;
        const int fX = ((s * 2 + 1) * 4 + mt) * 2 + kb;
        aG = *reinterpret_cast<const v16bf*>(frag + ((size_t)fG * 32 + lane) * 16);
        aX = *reinterpret_cast<const v16bf*>(frag + ((size_t)fX * 32 + lane) * 16);
      } else {
        const int row = mt * 16 + (lane & 15);
        const int ks  = kb * 32 + hi * 8;
        #pragma unroll
        for (int e = 0; e < 16; ++e) {
          const int k = ks + ((e < 8) ? e : (e + 8));
          aG[e] = (__bf16)Wg_s[row * CDIM + k];
          aX[e] = (__bf16)Wx_s[row * CDIM + k];
        }
      }
      acc = __builtin_amdgcn_wmma_f32_16x16x32_bf16(false, aG, false, bGf[kb],
                                                    (short)0, acc, false, false);
      acc = __builtin_amdgcn_wmma_f32_16x16x32_bf16(false, aX, false, bXf[kb],
                                                    (short)0, acc, false, false);
    }
    #pragma unroll
    for (int r = 0; r < 8; ++r) {
      const int o = mt * 16 + hi * 8 + r;
      float v = acc[r];
      v = v > 0.0f ? v : 0.0f;              // relu
      S += v * wp_s[o];
    }
  }
  S += __shfl_xor(S, 16);                   // combine M-halves (wave32)
  const float psi = S + bps[s];
  const float att = 1.0f / (1.0f + __expf(-psi));
  if (lane < 16) sAtt[s][pn] = att;
  __syncthreads();

  // -------- Phase 3: scale fp32 subbands, inverse Haar, stream out (b128) --------
  #pragma unroll
  for (int i = 0; i < 4; ++i) {
    const int idx = i * 256 + tid;
    const int c  = idx >> 4;
    const int pp = idx & 15;
    const int p0 = 2 * pp, p1 = 2 * pp + 1;
    const float ll0 = sSub[0][p0][c] * sAtt[0][p0];
    const float lh0 = sSub[1][p0][c] * sAtt[1][p0];
    const float hl0 = sSub[2][p0][c] * sAtt[2][p0];
    const float hh0 = sSub[3][p0][c] * sAtt[3][p0];
    const float ll1 = sSub[0][p1][c] * sAtt[0][p1];
    const float lh1 = sSub[1][p1][c] * sAtt[1][p1];
    const float hl1 = sSub[2][p1][c] * sAtt[2][p1];
    const float hh1 = sSub[3][p1][c] * sAtt[3][p1];
    const float4 r0 = make_float4((ll0 + lh0 + hl0 + hh0) * 0.5f,
                                  (ll0 + lh0 - hl0 - hh0) * 0.5f,
                                  (ll1 + lh1 + hl1 + hh1) * 0.5f,
                                  (ll1 + lh1 - hl1 - hh1) * 0.5f);
    const float4 r1 = make_float4((ll0 - lh0 + hl0 - hh0) * 0.5f,
                                  (ll0 - lh0 - hl0 + hh0) * 0.5f,
                                  (ll1 - lh1 + hl1 - hh1) * 0.5f,
                                  (ll1 - lh1 - hl1 + hh1) * 0.5f);
    const long base = (((long)(b * CDIM + c) * HDIM + 2 * h2) * WDIM) + (w0 + 4 * pp);
    *reinterpret_cast<float4*>(out + base)        = r0;
    *reinterpret_cast<float4*>(out + base + WDIM) = r1;
  }
}

extern "C" void kernel_launch(void* const* d_in, const int* in_sizes, int n_in,
                              void* d_out, int out_size, void* d_ws, size_t ws_size,
                              hipStream_t stream) {
  (void)in_sizes; (void)n_in; (void)out_size;
  const float* xe  = (const float*)d_in[0];
  const float* xd  = (const float*)d_in[1];
  const float* Wg  = (const float*)d_in[2];
  const float* bg  = (const float*)d_in[3];
  const float* Wx  = (const float*)d_in[4];
  const float* bx  = (const float*)d_in[5];
  const float* Wps = (const float*)d_in[6];
  const float* bps = (const float*)d_in[7];
  float* out = (float*)d_out;

  const int nblocks = BDIM * H2 * (W2 / PTILE);   // 8*128*4 = 4096 workgroups

  if (ws_size >= (size_t)WS_TOTAL) {
    prep_weights<<<dim3(8), dim3(256), 0, stream>>>(Wg, Wx, bg, bx, d_ws);
    wavelet_attn_gate_fused<true><<<dim3(nblocks), dim3(256), 0, stream>>>(
        xe, xd, Wg, bg, Wx, bx, Wps, bps, d_ws, out);
  } else {
    wavelet_attn_gate_fused<false><<<dim3(nblocks), dim3(256), 0, stream>>>(
        xe, xd, Wg, bg, Wx, bx, Wps, bps, d_ws, out);
  }
}
